// Net_86131274154323
// MI455X (gfx1250) — matrix operand order repaired and checked by reading, not compiled, not run on hardware
//
#include <hip/hip_runtime.h>

// ---------- types ----------
typedef __attribute__((ext_vector_type(16))) __bf16        v16bf;
typedef __attribute__((ext_vector_type(8)))  float         v8f;
typedef __attribute__((ext_vector_type(4)))  unsigned int  u32x4;

union FragU { u32x4 q[2]; v16bf v; };

__device__ __forceinline__ unsigned short f2bf(float f) {
    unsigned int u = __builtin_bit_cast(unsigned int, f);
    unsigned int r = u + 0x7FFFu + ((u >> 16) & 1u);   // RNE
    return (unsigned short)(r >> 16);
}
__device__ __forceinline__ float bf2f(unsigned short h) {
    unsigned int u = ((unsigned int)h) << 16;
    return __builtin_bit_cast(float, u);
}
__device__ __forceinline__ unsigned short vget16(u32x4 v, int e) {
    unsigned int d = v[e >> 1];
    return (unsigned short)((e & 1) ? (d >> 16) : (d & 0xffffu));
}

// CDNA5 async DMA: global -> LDS, 16B per lane, tracked by ASYNCcnt (GVS mode).
__device__ __forceinline__ void async_b128(unsigned lds, unsigned goff, const void* base) {
    asm volatile("global_load_async_to_lds_b128 %0, %1, %2"
                 :: "v"(lds), "v"(goff), "s"(base)
                 : "memory");
}
__device__ __forceinline__ void wait_async0() {
    asm volatile("s_wait_asynccnt 0x0" ::: "memory");
}
__device__ __forceinline__ unsigned lds_addr_of(const void* p) {
    return (unsigned)(size_t)p;   // flat LDS address truncates to wave-relative LDS offset
}

// ---------- deterministic abs-mean threshold (two stage) ----------
__global__ __launch_bounds__(256) void absum_part(const float* __restrict__ w, int n,
                                                  float* __restrict__ partial) {
    __shared__ float red[256];
    int tid = threadIdx.x;
    float s = 0.f;
    for (int i = blockIdx.x * 256 + tid; i < n; i += 256 * 256) s += fabsf(w[i]);
    red[tid] = s; __syncthreads();
    for (int off = 128; off; off >>= 1) {
        if (tid < off) red[tid] += red[tid + off];
        __syncthreads();
    }
    if (tid == 0) partial[blockIdx.x] = red[0];
}
__global__ __launch_bounds__(256) void absum_fin(const float* __restrict__ partial, float n,
                                                 float* __restrict__ thr) {
    __shared__ float red[256];
    int tid = threadIdx.x;
    red[tid] = partial[tid]; __syncthreads();
    for (int off = 128; off; off >>= 1) {
        if (tid < off) red[tid] += red[tid + off];
        __syncthreads();
    }
    if (tid == 0) *thr = 0.7f * red[0] / n;
}

// ---------- weight packing ----------
__global__ __launch_bounds__(256) void pack_conv_k(const float* __restrict__ w,
                                                   const float* __restrict__ thr,
                                                   unsigned short* __restrict__ Wp,
                                                   int Co, int Ci) {
    int idx = blockIdx.x * 256 + threadIdx.x;
    int tot = Co * 9 * Ci;
    if (idx >= tot) return;
    int n = idx / (9 * Ci);
    int r = idx - n * 9 * Ci;
    int t = r / Ci;
    int ci = r - t * Ci;
    int ky = t / 3, kx = t - 3 * ky;
    float v = w[(((size_t)n * Ci + ci) * 3 + ky) * 3 + kx];
    float d = *thr;
    float q = (v > d) ? 1.f : ((v < -d) ? -1.f : 0.f);
    Wp[idx] = f2bf(q);
}
__global__ __launch_bounds__(256) void pack_w0_k(const float* __restrict__ w,
                                                 unsigned short* __restrict__ Wp) {
    int idx = blockIdx.x * 256 + threadIdx.x;
    if (idx >= 384 * 32) return;
    int n = idx >> 5, k = idx & 31;
    unsigned short out = 0;
    if (k < 27) {
        int t = k / 3, c = k - 3 * t;
        int ky = t / 3, kx = t - 3 * ky;
        out = f2bf(w[(((size_t)n * 3 + c) * 3 + ky) * 3 + kx]);
    }
    Wp[idx] = out;
}
__global__ __launch_bounds__(256) void pack_fc1_k(const float* __restrict__ w,
                                                  const float* __restrict__ thr,
                                                  unsigned short* __restrict__ Wp) {
    int idx = blockIdx.x * 256 + threadIdx.x;
    const int K = 24576;
    if (idx >= 1024 * K) return;
    int n = idx / K;
    int k = idx - n * K;
    int s = k / 1536;
    int c = k - s * 1536;
    float v = w[(size_t)n * K + c * 16 + s];
    float d = *thr;
    float q = (v > d) ? 1.f : ((v < -d) ? -1.f : 0.f);
    Wp[idx] = f2bf(q);
}

// ---------- conv0: Cin=3, single K=32 chunk, fused ht -> BN -> celu ----------
__global__ __launch_bounds__(256) void conv0_k(const float* __restrict__ x,
                                               const unsigned short* __restrict__ Wp0,
                                               const float* __restrict__ g, const float* __restrict__ bt,
                                               const float* __restrict__ mu, const float* __restrict__ var,
                                               unsigned short* __restrict__ out) {
    __shared__ unsigned short As[128 * 40];
    __shared__ unsigned short Bs[64 * 40];
    const int tid  = threadIdx.x;
    const int wave = tid >> 5, lane = tid & 31;
    const int hlf  = lane >> 4, l16 = lane & 15;
    const int mbase = blockIdx.x * 128;
    const int nbase = blockIdx.y * 64;

    {
        const int rA = tid >> 1, hA = tid & 1;
        const int p = mbase + rA;
        const int b = p >> 10, rem = p & 1023;
        const int h = rem >> 5, w = rem & 31;
        for (int e = 0; e < 16; ++e) {
            int k = hA * 16 + e;
            unsigned short vbf = 0;
            if (k < 27) {
                int t = k / 3, cc = k - 3 * t;
                int dy = t / 3 - 1, dx = t % 3 - 1;
                int hs = h + dy, wss = w + dx;
                if ((unsigned)hs < 32u && (unsigned)wss < 32u)
                    vbf = f2bf(x[(((size_t)b * 3 + cc) * 32 + hs) * 32 + wss]);
            }
            As[rA * 40 + hA * 16 + e] = vbf;
        }
        const int nB = tid >> 2, qB = tid & 3;
        *(u32x4*)&Bs[nB * 40 + qB * 8] = *(const u32x4*)(Wp0 + (size_t)(nbase + nB) * 32 + qB * 8);
    }
    __syncthreads();

    v8f acc[4];
    for (int j = 0; j < 4; ++j)
        for (int e = 0; e < 8; ++e) acc[j][e] = 0.f;

    FragU fa;
    fa.q[0] = *(const u32x4*)&As[(16 * wave + l16) * 40 + 8 * hlf];
    fa.q[1] = *(const u32x4*)&As[(16 * wave + l16) * 40 + 16 + 8 * hlf];
    for (int j = 0; j < 4; ++j) {
        FragU fb;
        const int brow = (16 * j + l16) * 40 + 8 * hlf;
        fb.q[0] = *(const u32x4*)&Bs[brow];
        fb.q[1] = *(const u32x4*)&Bs[brow + 16];
        acc[j] = __builtin_amdgcn_wmma_f32_16x16x32_bf16(false, fa.v, false, fb.v,
                                                         (short)0, acc[j], false, false);
    }

    const int mloc = 16 * wave + 8 * hlf;
    for (int j = 0; j < 4; ++j) {
        const int n = nbase + 16 * j + l16;
        const float scale = g[n] * rsqrtf(var[n] + 1e-5f);
        const float mm = mu[n], bb = bt[n];
        for (int v = 0; v < 8; ++v) {
            float val = acc[j][v];
            val = fminf(fmaxf(val, -1.f), 1.f);           // hardtanh
            val = (val - mm) * scale + bb;                // BN
            val = val > 0.f ? val : (expf(val) - 1.f);    // celu
            out[(size_t)(mbase + mloc + v) * 384 + n] = f2bf(val);
        }
    }
}

// ---------- generic 3x3 pad-1 implicit-GEMM conv, NHWC bf16 ----------
// tile M=128 x N=128, 8 wave32/block, K chunks of 32, async double-buffered LDS
template <int FUSE_HT>
__global__ __launch_bounds__(256) void conv3x3_k(const unsigned short* __restrict__ act,
                                                 const unsigned short* __restrict__ Wp,
                                                 unsigned short* __restrict__ out,
                                                 int H, int W, int Cin, int Cout) {
    __shared__ unsigned short As[2][128 * 40];
    __shared__ unsigned short Bs[2][128 * 40];
    const int tid  = threadIdx.x;
    const int wave = tid >> 5, lane = tid & 31;
    const int hlf  = lane >> 4, l16 = lane & 15;
    const int mbase = blockIdx.x * 128;
    const int nbase = blockIdx.y * 128;
    const int Kr = 9 * Cin;
    const int rA = tid >> 1, hA = tid & 1;   // A/B staging: 2 threads per row, 32B each
    const int p = mbase + rA;
    const int HW = H * W;
    const int bImg = p / HW;
    const int rem  = p - bImg * HW;
    const int hPix = rem / W;
    const int wPix = rem - hPix * W;

    const unsigned ldsA0 = lds_addr_of(&As[0][rA * 40 + hA * 16]);
    const unsigned ldsB0 = lds_addr_of(&Bs[0][rA * 40 + hA * 16]);
    const unsigned strideA = 128u * 40u * 2u;
    const unsigned strideB = 128u * 40u * 2u;

    const int nChunks = Cin >> 5;
    const int total = 9 * nChunks;

    auto issue = [&](int q, int buf) {
        const int t  = q / nChunks;
        const int c  = q - t * nChunks;
        const int cb = c << 5;
        const int dy = t / 3 - 1, dx = t % 3 - 1;
        const int hs = hPix + dy, wss = wPix + dx;
        const unsigned la = ldsA0 + (unsigned)buf * strideA;
        if (((unsigned)hs < (unsigned)H) && ((unsigned)wss < (unsigned)W)) {
            const unsigned ga =
                (unsigned)((((bImg * H + hs) * W + wss) * Cin + cb + hA * 16) * 2);
            async_b128(la, ga, act);
            async_b128(la + 16, ga + 16, act);
        } else {
            u32x4 z = {0, 0, 0, 0};   // zero-fill halo via normal LDS store
            *(u32x4*)&As[buf][rA * 40 + hA * 16]     = z;
            *(u32x4*)&As[buf][rA * 40 + hA * 16 + 8] = z;
        }
        const unsigned gb =
            (unsigned)(((size_t)(nbase + rA) * Kr + (size_t)t * Cin + cb + hA * 16) * 2);
        const unsigned lb = ldsB0 + (unsigned)buf * strideB;
        async_b128(lb, gb, Wp);
        async_b128(lb + 16, gb + 16, Wp);
    };

    v8f acc[8];
    for (int j = 0; j < 8; ++j)
        for (int e = 0; e < 8; ++e) acc[j][e] = 0.f;

    issue(0, 0);
    for (int q = 0; q < total; ++q) {
        const int buf = q & 1;
        wait_async0();        // own wave's chunk-q DMA done
        __syncthreads();      // everyone's chunk-q data visible; prev compute finished
        if (q + 1 < total) issue(q + 1, buf ^ 1);   // overlaps this chunk's WMMAs

        FragU fa;   // A fragment per ISA 16-bit A 16x32 lane map
        fa.q[0] = *(const u32x4*)&As[buf][(16 * wave + l16) * 40 + 8 * hlf];
        fa.q[1] = *(const u32x4*)&As[buf][(16 * wave + l16) * 40 + 16 + 8 * hlf];
        for (int j = 0; j < 8; ++j) {
            FragU fb;   // B fragment: lane -> N=l16, K = 8*hlf + e (e<8), 16 + 8*hlf + (e-8) (e>=8)
            const int brow = (16 * j + l16) * 40 + 8 * hlf;
            fb.q[0] = *(const u32x4*)&Bs[buf][brow];
            fb.q[1] = *(const u32x4*)&Bs[buf][brow + 16];
            acc[j] = __builtin_amdgcn_wmma_f32_16x16x32_bf16(false, fa.v, false, fb.v,
                                                             (short)0, acc[j], false, false);
        }
    }

    const int mloc = 16 * wave + 8 * hlf;
    for (int j = 0; j < 8; ++j) {
        const int n = nbase + 16 * j + l16;
        for (int v = 0; v < 8; ++v) {
            float val = acc[j][v];
            if (FUSE_HT) val = fminf(fmaxf(val, -1.f), 1.f);
            out[(size_t)(mbase + mloc + v) * Cout + n] = f2bf(val);
        }
    }
}

// ---------- fused 2x2 maxpool + hardtanh, NHWC bf16 ----------
__global__ __launch_bounds__(256) void pool_ht_k(const unsigned short* __restrict__ in,
                                                 unsigned short* __restrict__ out,
                                                 int B, int H, int W, int C) {
    const int C8 = C >> 3, Wo = W >> 1, Ho = H >> 1;
    const int total = B * Ho * Wo * C8;
    int idx = blockIdx.x * 256 + threadIdx.x;
    if (idx >= total) return;
    int c8 = idx % C8;
    int r  = idx / C8;
    int wo = r % Wo; r /= Wo;
    int ho = r % Ho;
    int b  = r / Ho;
    size_t base = ((size_t)(b * H + 2 * ho) * W + 2 * wo) * C + (size_t)c8 * 8;
    const u32x4 v00 = *(const u32x4*)(in + base);
    const u32x4 v01 = *(const u32x4*)(in + base + C);
    const u32x4 v10 = *(const u32x4*)(in + base + (size_t)W * C);
    const u32x4 v11 = *(const u32x4*)(in + base + (size_t)W * C + C);
    u32x4 o = {0, 0, 0, 0};
    for (int e = 0; e < 8; ++e) {
        float m = fmaxf(fmaxf(bf2f(vget16(v00, e)), bf2f(vget16(v01, e))),
                        fmaxf(bf2f(vget16(v10, e)), bf2f(vget16(v11, e))));
        m = fminf(fmaxf(m, -1.f), 1.f);
        unsigned int h = f2bf(m);
        o[e >> 1] |= h << ((e & 1) * 16);
    }
    *(u32x4*)(out + (size_t)idx * 8) = o;
}

// ---------- fc1: M=128 GEMM (K=24576, N tile 128), async double-buffered, BN+ht ----------
__global__ __launch_bounds__(256) void fc1_k(const unsigned short* __restrict__ act,
                                             const unsigned short* __restrict__ Wp,
                                             const float* __restrict__ g, const float* __restrict__ bt,
                                             const float* __restrict__ mu, const float* __restrict__ var,
                                             unsigned short* __restrict__ out) {
    __shared__ unsigned short As[2][128 * 40];
    __shared__ unsigned short Bs[2][128 * 40];
    const int tid  = threadIdx.x;
    const int wave = tid >> 5, lane = tid & 31;
    const int hlf  = lane >> 4, l16 = lane & 15;
    const int nbase = blockIdx.x * 128;
    const int K = 24576;
    const int rA = tid >> 1, hA = tid & 1;

    const unsigned ldsA0 = lds_addr_of(&As[0][rA * 40 + hA * 16]);
    const unsigned ldsB0 = lds_addr_of(&Bs[0][rA * 40 + hA * 16]);
    const unsigned strideA = 128u * 40u * 2u;
    const unsigned strideB = 128u * 40u * 2u;

    auto issue = [&](int q, int buf) {
        const int cb = q << 5;
        const unsigned ga = (unsigned)(((size_t)rA * K + cb + hA * 16) * 2);
        const unsigned la = ldsA0 + (unsigned)buf * strideA;
        async_b128(la, ga, act);
        async_b128(la + 16, ga + 16, act);
        const unsigned gb = (unsigned)(((size_t)(nbase + rA) * K + cb + hA * 16) * 2);
        const unsigned lb = ldsB0 + (unsigned)buf * strideB;
        async_b128(lb, gb, Wp);
        async_b128(lb + 16, gb + 16, Wp);
    };

    v8f acc[8];
    for (int j = 0; j < 8; ++j)
        for (int e = 0; e < 8; ++e) acc[j][e] = 0.f;

    const int total = K / 32;
    issue(0, 0);
    for (int q = 0; q < total; ++q) {
        const int buf = q & 1;
        wait_async0();
        __syncthreads();
        if (q + 1 < total) issue(q + 1, buf ^ 1);
        FragU fa;
        fa.q[0] = *(const u32x4*)&As[buf][(16 * wave + l16) * 40 + 8 * hlf];
        fa.q[1] = *(const u32x4*)&As[buf][(16 * wave + l16) * 40 + 16 + 8 * hlf];
        for (int j = 0; j < 8; ++j) {
            FragU fb;
            const int brow = (16 * j + l16) * 40 + 8 * hlf;
            fb.q[0] = *(const u32x4*)&Bs[buf][brow];
            fb.q[1] = *(const u32x4*)&Bs[buf][brow + 16];
            acc[j] = __builtin_amdgcn_wmma_f32_16x16x32_bf16(false, fa.v, false, fb.v,
                                                             (short)0, acc[j], false, false);
        }
    }

    const int mloc = 16 * wave + 8 * hlf;
    for (int j = 0; j < 8; ++j) {
        const int n = nbase + 16 * j + l16;
        const float scale = g[n] * rsqrtf(var[n] + 1e-5f);
        const float mm = mu[n], bb = bt[n];
        for (int v = 0; v < 8; ++v) {
            float val = (acc[j][v] - mm) * scale + bb;
            val = fminf(fmaxf(val, -1.f), 1.f);
            out[(size_t)(mloc + v) * 1024 + n] = f2bf(val);
        }
    }
}

// ---------- fc2 (ternary, inline) + log_softmax ----------
__global__ __launch_bounds__(128) void fc2_k(const unsigned short* __restrict__ act,
                                             const float* __restrict__ w,
                                             const float* __restrict__ thr,
                                             float* __restrict__ out) {
    __shared__ float red[10 * 128];
    const int b = blockIdx.x, tid = threadIdx.x;
    const float d = *thr;
    float acc[10];
    for (int o = 0; o < 10; ++o) acc[o] = 0.f;
    for (int k = tid; k < 1024; k += 128) {
        float a = bf2f(act[(size_t)b * 1024 + k]);
        for (int o = 0; o < 10; ++o) {
            float v = w[o * 1024 + k];
            float q = (v > d) ? 1.f : ((v < -d) ? -1.f : 0.f);
            acc[o] += a * q;
        }
    }
    for (int o = 0; o < 10; ++o) red[o * 128 + tid] = acc[o];
    __syncthreads();
    for (int off = 64; off; off >>= 1) {
        if (tid < off)
            for (int o = 0; o < 10; ++o) red[o * 128 + tid] += red[o * 128 + tid + off];
        __syncthreads();
    }
    if (tid == 0) {
        float mx = -1e30f;
        float logits[10];
        for (int o = 0; o < 10; ++o) { logits[o] = red[o * 128]; mx = fmaxf(mx, logits[o]); }
        float s = 0.f;
        for (int o = 0; o < 10; ++o) s += expf(logits[o] - mx);
        float lse = logf(s) + mx;
        for (int o = 0; o < 10; ++o) out[b * 10 + o] = logits[o] - lse;
    }
}

// ---------- host launcher ----------
extern "C" void kernel_launch(void* const* d_in, const int* in_sizes, int n_in,
                              void* d_out, int out_size, void* d_ws, size_t ws_size,
                              hipStream_t stream) {
    (void)in_sizes; (void)n_in; (void)out_size; (void)ws_size;
    const float* x     = (const float*)d_in[0];
    const float* w0    = (const float*)d_in[1];
    const float* bn0g  = (const float*)d_in[2];
    const float* bn0b  = (const float*)d_in[3];
    const float* bn0m  = (const float*)d_in[4];
    const float* bn0v  = (const float*)d_in[5];
    const float* w1    = (const float*)d_in[6];
    const float* w2    = (const float*)d_in[7];
    const float* w3    = (const float*)d_in[8];
    const float* w4    = (const float*)d_in[9];
    const float* w5    = (const float*)d_in[10];
    const float* fc1w  = (const float*)d_in[11];
    const float* bn1g  = (const float*)d_in[12];
    const float* bn1b  = (const float*)d_in[13];
    const float* bn1m  = (const float*)d_in[14];
    const float* bn1v  = (const float*)d_in[15];
    const float* fc2w  = (const float*)d_in[16];

    char* ws = (char*)d_ws;
    size_t off = 0;
    auto alloc = [&](size_t bytes) -> char* {
        char* p = ws + off;
        off += (bytes + 255) & ~(size_t)255;
        return p;
    };
    float* thr      = (float*)alloc(64);
    float* partial  = (float*)alloc(256 * sizeof(float));
    unsigned short* Wp0  = (unsigned short*)alloc((size_t)384 * 32 * 2);
    unsigned short* Wp1  = (unsigned short*)alloc((size_t)384 * 3456 * 2);
    unsigned short* Wp2  = (unsigned short*)alloc((size_t)768 * 3456 * 2);
    unsigned short* Wp3  = (unsigned short*)alloc((size_t)768 * 6912 * 2);
    unsigned short* Wp4  = (unsigned short*)alloc((size_t)1536 * 6912 * 2);
    unsigned short* Wp5  = (unsigned short*)alloc((size_t)1536 * 13824 * 2);
    unsigned short* Wfc1 = (unsigned short*)alloc((size_t)1024 * 24576 * 2);
    unsigned short* bufA = (unsigned short*)alloc((size_t)128 * 32 * 32 * 384 * 2);
    unsigned short* bufB = (unsigned short*)alloc((size_t)128 * 32 * 32 * 384 * 2);

    auto reduceThr = [&](const float* w, int n, float* t) {
        absum_part<<<256, 256, 0, stream>>>(w, n, partial);
        absum_fin<<<1, 256, 0, stream>>>(partial, (float)n, t);
    };
    reduceThr(w1,   384 * 384 * 9,   thr + 0);
    reduceThr(w2,   768 * 384 * 9,   thr + 1);
    reduceThr(w3,   768 * 768 * 9,   thr + 2);
    reduceThr(w4,   1536 * 768 * 9,  thr + 3);
    reduceThr(w5,   1536 * 1536 * 9, thr + 4);
    reduceThr(fc1w, 1024 * 24576,    thr + 5);
    reduceThr(fc2w, 10 * 1024,       thr + 6);

    auto cdiv = [](int a, int b) { return (a + b - 1) / b; };
    pack_w0_k<<<cdiv(384 * 32, 256), 256, 0, stream>>>(w0, Wp0);
    pack_conv_k<<<cdiv(384 * 9 * 384, 256), 256, 0, stream>>>(w1, thr + 0, Wp1, 384, 384);
    pack_conv_k<<<cdiv(768 * 9 * 384, 256), 256, 0, stream>>>(w2, thr + 1, Wp2, 768, 384);
    pack_conv_k<<<cdiv(768 * 9 * 768, 256), 256, 0, stream>>>(w3, thr + 2, Wp3, 768, 768);
    pack_conv_k<<<cdiv(1536 * 9 * 768, 256), 256, 0, stream>>>(w4, thr + 3, Wp4, 1536, 768);
    pack_conv_k<<<cdiv(1536 * 9 * 1536, 256), 256, 0, stream>>>(w5, thr + 4, Wp5, 1536, 1536);
    pack_fc1_k<<<cdiv(1024 * 24576, 256), 256, 0, stream>>>(fc1w, thr + 5, Wfc1);

    conv0_k<<<dim3(131072 / 128, 384 / 64), 256, 0, stream>>>(x, Wp0, bn0g, bn0b, bn0m, bn0v, bufA);
    conv3x3_k<0><<<dim3(131072 / 128, 384 / 128), 256, 0, stream>>>(bufA, Wp1, bufB, 32, 32, 384, 384);
    pool_ht_k<<<cdiv(128 * 16 * 16 * 384 / 8, 256), 256, 0, stream>>>(bufB, bufA, 128, 32, 32, 384);
    conv3x3_k<1><<<dim3(32768 / 128, 768 / 128), 256, 0, stream>>>(bufA, Wp2, bufB, 16, 16, 384, 768);
    conv3x3_k<0><<<dim3(32768 / 128, 768 / 128), 256, 0, stream>>>(bufB, Wp3, bufA, 16, 16, 768, 768);
    pool_ht_k<<<cdiv(128 * 8 * 8 * 768 / 8, 256), 256, 0, stream>>>(bufA, bufB, 128, 16, 16, 768);
    conv3x3_k<1><<<dim3(8192 / 128, 1536 / 128), 256, 0, stream>>>(bufB, Wp4, bufA, 8, 8, 768, 1536);
    conv3x3_k<0><<<dim3(8192 / 128, 1536 / 128), 256, 0, stream>>>(bufA, Wp5, bufB, 8, 8, 1536, 1536);
    pool_ht_k<<<cdiv(128 * 4 * 4 * 1536 / 8, 256), 256, 0, stream>>>(bufB, bufA, 128, 8, 8, 1536);
    fc1_k<<<dim3(1024 / 128), 256, 0, stream>>>(bufA, Wfc1, bn1g, bn1b, bn1m, bn1v, bufB);
    fc2_k<<<128, 128, 0, stream>>>(bufB, fc2w, thr + 6, (float*)d_out);
}
